// LSTMEncoder_31860067402111
// MI455X (gfx1250) — compile-verified, matrix operand order checked by reading, not run on hardware
//
#include <hip/hip_runtime.h>
#include <math.h>

#define VOCABN 32000
#define EMBN   512
#define HIDN   1024
#define VQDN   512
#define NEMBN  1024
#define BBN    32
#define TTN    512
#define NWG    32
#define USLICE 32   // hidden units per workgroup in LSTM kernel

typedef __attribute__((ext_vector_type(16))) __bf16 v16bf;
typedef __attribute__((ext_vector_type(8)))  float  v8f;
typedef __attribute__((ext_vector_type(4)))  unsigned u32x4;
typedef __attribute__((ext_vector_type(8)))  int      i32x8;
typedef __attribute__((ext_vector_type(4)))  int      i32x4;

// ---------------- bf16 helpers ----------------
__device__ __forceinline__ unsigned short f2bf(float f) {
  unsigned int u = __builtin_bit_cast(unsigned int, f);
  unsigned int r = u + 0x7FFFu + ((u >> 16) & 1u);   // round-to-nearest-even
  return (unsigned short)(r >> 16);
}
__device__ __forceinline__ __bf16 bfbits(unsigned short s) {
  return __builtin_bit_cast(__bf16, s);
}

// ---------------- WMMA fragment load (CDNA5 ISA 7.12.2) -----------------
// 16-bit A-matrix 16x32: lane group g = lane/16, element e holds
//   K = k0 + e + 8*g + (e>=8 ? 8 : 0), row M = m0 + (lane&15)
__device__ __forceinline__ v16bf load_a_frag(const unsigned short* __restrict__ A,
                                             int lda, int m0, int k0) {
  int lane = threadIdx.x & 31;
  int g = lane >> 4;
  const unsigned short* p = A + (size_t)(m0 + (lane & 15)) * lda + k0 + 8 * g;
  v16bf a;
#pragma unroll
  for (int e = 0; e < 16; ++e) a[e] = bfbits(p[e + (e >= 8 ? 8 : 0)]);
  return a;
}
// B matrix 32x16 (K x N), sourced from row-major Bm[N][K] (weights layout):
// lane holds column n = n0 + (lane&15); element e holds same K pattern as A.
__device__ __forceinline__ v16bf load_bT_frag(const unsigned short* __restrict__ Bm,
                                              int ldb, int n0, int k0) {
  int lane = threadIdx.x & 31;
  int g = lane >> 4;
  const unsigned short* p = Bm + (size_t)(n0 + (lane & 15)) * ldb + k0 + 8 * g;
  v16bf b;
#pragma unroll
  for (int e = 0; e < 16; ++e) b[e] = bfbits(p[e + (e >= 8 ? 8 : 0)]);
  return b;
}
__device__ __forceinline__ v8f wmma_bf16(v16bf a, v16bf b, v8f c) {
  return __builtin_amdgcn_wmma_f32_16x16x32_bf16(false, a, false, b, (short)0, c,
                                                 false, false);
}

// ---------------- Tensor Data Mover: 2D tile -> LDS --------------------------
// Builds D# per CDNA5 ISA and issues
// tensor_load_to_lds (TENSORcnt-tracked).  All arguments wave-uniform.
__device__ __forceinline__ void tdm_load_2d(const unsigned short* gbase,
                                            unsigned lds_off, int tile_k,
                                            int tile_rows, int stride_elems,
                                            int tensor_d0, int tensor_d1) {
  unsigned long long ga = (unsigned long long)(size_t)gbase;
  u32x4 g0;
  g0[0] = 1u;                                            // count=1
  g0[1] = lds_off;                                       // lds_addr (bytes)
  g0[2] = (unsigned)(ga & 0xFFFFFFFFu);                  // global_addr[31:0]
  g0[3] = (unsigned)((ga >> 32) & 0x01FFFFFFu) | (2u << 30); // addr hi | type=2
  i32x8 g1;
  g1[0] = (int)(1u << 16);                               // data_size=1 (2B)
  g1[1] = (int)(((unsigned)tensor_d0 & 0xFFFFu) << 16);  // tensor_dim0 lo16
  g1[2] = (int)((((unsigned)tensor_d0 >> 16) & 0xFFFFu) |
                (((unsigned)tensor_d1 & 0xFFFFu) << 16)); // d0 hi | d1 lo
  g1[3] = (int)((((unsigned)tensor_d1 >> 16) & 0xFFFFu) |
                ((unsigned)tile_k << 16));               // d1 hi | tile_dim0
  g1[4] = (int)((unsigned)tile_rows & 0xFFFFu);          // tile_dim1 (dim2=0)
  g1[5] = stride_elems;                                  // dim0_stride lo32
  g1[6] = 0;                                             // stride hi, dim1_stride lo
  g1[7] = 0;                                             // dim1_stride hi
  i32x4 z4 = {0, 0, 0, 0};
#if __clang_major__ >= 23
  i32x8 z8 = {0, 0, 0, 0, 0, 0, 0, 0};
  __builtin_amdgcn_tensor_load_to_lds(g0, g1, z4, z4, z8, 0);
#else
  __builtin_amdgcn_tensor_load_to_lds(g0, g1, z4, z4, 0);
#endif
}
__device__ __forceinline__ unsigned lds_offset_of(const void* p) {
  return (unsigned)(size_t)p;   // LDS aperture: low 32 bits = LDS byte offset
}

// ---------------- small prep kernels ----------------
__global__ void embed_kernel(const int* __restrict__ idx,
                             const float* __restrict__ ew,
                             unsigned short* __restrict__ x, size_t n) {
  size_t i = (size_t)blockIdx.x * blockDim.x + threadIdx.x;
  if (i >= n) return;
  size_t bt = i / EMBN;
  int e = (int)(i % EMBN);
  x[i] = f2bf(ew[(size_t)idx[bt] * EMBN + e]);
}

__global__ void f32_to_bf16_kernel(const float* __restrict__ s,
                                   unsigned short* __restrict__ d, size_t n) {
  size_t i = (size_t)blockIdx.x * blockDim.x + threadIdx.x;
  if (i < n) d[i] = f2bf(s[i]);
}

__global__ void bias_kernel(const float* __restrict__ b_ih,
                            const float* __restrict__ b_hh,
                            float* __restrict__ bias) {
  int i = blockIdx.x * blockDim.x + threadIdx.x;
  if (i < 4 * HIDN) bias[i] = b_ih[i] + b_hh[i];
}

// codebook is [VQD, NEMB]; build bf16 transpose [NEMB][VQD]
__global__ void cbT_kernel(const float* __restrict__ cb,
                           unsigned short* __restrict__ cbT) {
  size_t i = (size_t)blockIdx.x * blockDim.x + threadIdx.x;
  if (i >= (size_t)NEMBN * VQDN) return;
  int c = (int)(i / VQDN);
  int d = (int)(i % VQDN);
  cbT[i] = f2bf(cb[(size_t)d * NEMBN + c]);
}

__global__ void cbnorm_kernel(const float* __restrict__ cb,
                              float* __restrict__ nrm) {
  int c = blockIdx.x * blockDim.x + threadIdx.x;
  if (c >= NEMBN) return;
  float s = 0.f;
  for (int d = 0; d < VQDN; ++d) {
    float v = cb[(size_t)d * NEMBN + c];
    s += v * v;
  }
  nrm[c] = s;
}

__global__ void init_kernel(unsigned short* __restrict__ h_bf,
                            unsigned* __restrict__ cnt,
                            float* __restrict__ diff) {
  int i = blockIdx.x * blockDim.x + threadIdx.x;
  if (i < BBN * HIDN) h_bf[i] = 0;
  if (i == 0) { *cnt = 0u; *diff = 0.f; }
}

// ---------------- TDM + LDS double-buffered WMMA GEMM ------------------------
// C[M,N] = A[M,K] * B[N,K]^T (+bias).  Block 256 thr = 8 waves; block tile
// 64(M) x 128(N); K-step 32.  Wave 0 DMAs next A/B tiles into LDS with the
// Tensor Data Mover while all waves compute WMMAs from the resident buffer.
__global__ __launch_bounds__(256) void wmma_gemm_tdm(
    const unsigned short* __restrict__ A, const unsigned short* __restrict__ Bm,
    float* __restrict__ C, unsigned short* __restrict__ Cbf,
    const float* __restrict__ bias, int M, int N, int K, int lda, int ldb,
    int ldc) {
  __shared__ unsigned short At[2][64][32];    // 8 KB
  __shared__ unsigned short Bt[2][128][32];   // 16 KB
  const int wave = threadIdx.x >> 5;
  const int lane = threadIdx.x & 31;
  const int mblk = blockIdx.y * 64;
  const int nblk = blockIdx.x * 128;
  const int m0 = (wave >> 1) * 16;            // local row tile
  const int nb = (wave & 1) * 64;             // local col base (4 tiles)
  const int nK = K / 32;

  if (wave == 0) {
    tdm_load_2d(A + (size_t)mblk * lda, lds_offset_of(&At[0][0][0]), 32, 64,
                lda, K, M);
    tdm_load_2d(Bm + (size_t)nblk * ldb, lds_offset_of(&Bt[0][0][0]), 32, 128,
                ldb, K, N);
  }

  v8f acc[4] = {};
  for (int kk = 0; kk < nK; ++kk) {
    const int cur = kk & 1;
    if (wave == 0) {
      if (kk + 1 < nK) {
        const int k0 = (kk + 1) * 32;
        tdm_load_2d(A + (size_t)mblk * lda + k0,
                    lds_offset_of(&At[cur ^ 1][0][0]), 32, 64, lda, K, M);
        tdm_load_2d(Bm + (size_t)nblk * ldb + k0,
                    lds_offset_of(&Bt[cur ^ 1][0][0]), 32, 128, ldb, K, N);
        __builtin_amdgcn_s_wait_tensorcnt(2);  // tile kk resident; kk+1 in flight
      } else {
        __builtin_amdgcn_s_wait_tensorcnt(0);
      }
    }
    __syncthreads();
    v16bf a = load_a_frag(&At[cur][0][0], 32, m0, 0);
#pragma unroll
    for (int j = 0; j < 4; ++j) {
      v16bf b = load_bT_frag(&Bt[cur][0][0], 32, nb + j * 16, 0);
      acc[j] = wmma_bf16(a, b, acc[j]);
    }
    __syncthreads();
  }

  const int g = lane >> 4;
#pragma unroll
  for (int j = 0; j < 4; ++j) {
    int col = nblk + nb + j * 16 + (lane & 15);
    float bv = bias ? bias[col] : 0.f;
#pragma unroll
    for (int i = 0; i < 8; ++i) {
      int row = mblk + m0 + i + 8 * g;
      float v = acc[j][i] + bv;
      C[(size_t)row * ldc + col] = v;
      if (Cbf) Cbf[(size_t)row * ldc + col] = f2bf(v);
    }
  }
}

// ---------------- persistent LSTM kernel ----------------
__device__ __forceinline__ float sigmoidf_(float x) {
  return 1.0f / (1.0f + __expf(-x));
}

__device__ __forceinline__ void grid_barrier(unsigned* cnt, unsigned target) {
  __syncthreads();
  if (threadIdx.x == 0) {
    __threadfence();
    __hip_atomic_fetch_add(cnt, 1u, __ATOMIC_ACQ_REL, __HIP_MEMORY_SCOPE_AGENT);
    while (__hip_atomic_load(cnt, __ATOMIC_ACQUIRE, __HIP_MEMORY_SCOPE_AGENT) <
           target) {
      __builtin_amdgcn_s_sleep(1);
    }
  }
  __syncthreads();
}

// 32 workgroups; WG owns hidden units [wg*32, wg*32+32). Per step, each WG
// computes gates[32 batch, 128 cols] = h(t-1)[32,1024] x w_hh_slice^T via
// bf16 WMMA, adds the precomputed xp slice, applies activations, updates c
// (LDS-resident for all 512 steps) and publishes h in bf16; atomic-counter
// grid barrier per step.  h (64 KB) and w_hh (8 MB bf16) stay L2-resident.
__global__ __launch_bounds__(256) void lstm_kernel(
    const float* __restrict__ xp, const unsigned short* __restrict__ whh,
    unsigned short* __restrict__ h_bf, unsigned short* __restrict__ hs_bf,
    unsigned* __restrict__ sync_cnt) {
  __shared__ float c_lds[USLICE][BBN];        // 4 KB, persists across steps
  __shared__ float g_lds[4 * USLICE][BBN];    // 16 KB, [gate col][batch]

  const int wg = blockIdx.x;
  const int u0 = wg * USLICE;
  const int tid = threadIdx.x;
  const int wave = tid >> 5;
  const int lane = tid & 31;
  const int lg = lane >> 4;

  for (int p = tid; p < USLICE * BBN; p += blockDim.x)
    (&c_lds[0][0])[p] = 0.f;
  __syncthreads();

  const int nloc0 = wave * 16;        // local gate col base (0..127)
  const int gb = nloc0 >> 5;          // gate block 0..3 (i,f,g,o)
  const int un = nloc0 & 31;          // unit offset within slice
  const unsigned short* Bp = whh + (size_t)(gb * HIDN + u0 + un) * HIDN;

  for (int t = 0; t < TTN; ++t) {
    v8f acc0 = {};
    v8f acc1 = {};
    if (t > 0) {
      for (int k0 = 0; k0 < HIDN; k0 += 32) {
        v16bf b = load_bT_frag(Bp, HIDN, 0, k0);
        v16bf a0 = load_a_frag(h_bf, HIDN, 0, k0);
        v16bf a1 = load_a_frag(h_bf, HIDN, 16, k0);
        acc0 = wmma_bf16(a0, b, acc0);
        acc1 = wmma_bf16(a1, b, acc1);
      }
    }
    {
      const int col = nloc0 + (lane & 15);                 // local 0..127
      const int gcol = gb * HIDN + u0 + un + (lane & 15);  // global gate col
#pragma unroll
      for (int i = 0; i < 8; ++i) {
        int b0 = i + 8 * lg;
        int b1 = 16 + b0;
        g_lds[col][b0] = acc0[i] + xp[((size_t)b0 * TTN + t) * (4 * HIDN) + gcol];
        g_lds[col][b1] = acc1[i] + xp[((size_t)b1 * TTN + t) * (4 * HIDN) + gcol];
      }
    }
    __syncthreads();
    for (int p = tid; p < USLICE * BBN; p += blockDim.x) {
      int u = p & (USLICE - 1);
      int b = p >> 5;
      float ig = sigmoidf_(g_lds[u][b]);
      float fg = sigmoidf_(g_lds[USLICE + u][b]);
      float gg = tanhf(g_lds[2 * USLICE + u][b]);
      float og = sigmoidf_(g_lds[3 * USLICE + u][b]);
      float c = fg * c_lds[u][b] + ig * gg;
      c_lds[u][b] = c;
      float h = og * tanhf(c);
      unsigned short hb = f2bf(h);
      h_bf[(size_t)b * HIDN + u0 + u] = hb;
      hs_bf[((size_t)b * TTN + t) * HIDN + u0 + u] = hb;
    }
    grid_barrier(sync_cnt, (unsigned)(NWG * (t + 1)));
  }
}

// ---------------- VQ argmin + gather + diff ----------------
// one wave per row; score = ||cb_c||^2 - 2*dot  (||f||^2 constant per row)
__global__ __launch_bounds__(256) void vq_argmin_kernel(
    const float* __restrict__ dots, const float* __restrict__ cbnorm,
    const float* __restrict__ vqs, const float* __restrict__ cb,
    float* __restrict__ out, float* __restrict__ diff_acc) {
  int row = blockIdx.x * 8 + (threadIdx.x >> 5);
  int lane = threadIdx.x & 31;
  const float* dr = dots + (size_t)row * NEMBN;
  float best = INFINITY;
  int bidx = 0x7fffffff;
  for (int c = lane; c < NEMBN; c += 32) {
    float s = cbnorm[c] - 2.0f * dr[c];
    if (s < best || (s == best && c < bidx)) { best = s; bidx = c; }
  }
#pragma unroll
  for (int off = 16; off > 0; off >>= 1) {
    float ob = __shfl_down(best, off);
    int oi = __shfl_down(bidx, off);
    if (ob < best || (ob == best && oi < bidx)) { best = ob; bidx = oi; }
  }
  bidx = __shfl(bidx, 0);
  float ssum = 0.f;
  for (int d = lane; d < VQDN; d += 32) {
    float q = cb[(size_t)d * NEMBN + bidx];   // codebook.T[bidx][d]
    out[(size_t)row * VQDN + d] = q;
    float df = q - vqs[(size_t)row * VQDN + d];
    ssum += df * df;
  }
#pragma unroll
  for (int off = 16; off > 0; off >>= 1) ssum += __shfl_down(ssum, off);
  if (lane == 0) atomicAdd(diff_acc, ssum);
}

__global__ void finalize_kernel(const float* __restrict__ diff,
                                float* __restrict__ out) {
  if (threadIdx.x == 0 && blockIdx.x == 0)
    out[(size_t)BBN * TTN * VQDN] =
        *diff / (float)((size_t)BBN * TTN * VQDN);
}

// ---------------- host side ----------------
extern "C" void kernel_launch(void* const* d_in, const int* in_sizes, int n_in,
                              void* d_out, int out_size, void* d_ws,
                              size_t ws_size, hipStream_t stream) {
  (void)in_sizes; (void)n_in; (void)out_size; (void)ws_size;
  const int*   inputs   = (const int*)d_in[0];
  const float* embed_w  = (const float*)d_in[1];
  const float* w_ih     = (const float*)d_in[2];
  const float* w_hh     = (const float*)d_in[3];
  const float* b_ih     = (const float*)d_in[4];
  const float* b_hh     = (const float*)d_in[5];
  const float* lin_w    = (const float*)d_in[6];
  const float* codebook = (const float*)d_in[7];
  float* out = (float*)d_out;

  char* ws = (char*)d_ws;
  size_t off = 0;
  auto alloc = [&](size_t bytes) -> void* {
    void* p = ws + off;
    off += bytes;
    off = (off + 255) & ~(size_t)255;
    return p;
  };
  unsigned short* x_bf   = (unsigned short*)alloc((size_t)BBN * TTN * EMBN * 2);
  unsigned short* wih_bf = (unsigned short*)alloc((size_t)4 * HIDN * EMBN * 2);
  unsigned short* whh_bf = (unsigned short*)alloc((size_t)4 * HIDN * HIDN * 2);
  unsigned short* lin_bf = (unsigned short*)alloc((size_t)VQDN * HIDN * 2);
  unsigned short* cbT_bf = (unsigned short*)alloc((size_t)NEMBN * VQDN * 2);
  float* bias            = (float*)alloc((size_t)4 * HIDN * 4);
  float* cbnorm          = (float*)alloc((size_t)NEMBN * 4);
  float* xp              = (float*)alloc((size_t)BBN * TTN * 4 * HIDN * 4);
  unsigned short* hs_bf  = (unsigned short*)alloc((size_t)BBN * TTN * HIDN * 2);
  float* vqs             = (float*)alloc((size_t)BBN * TTN * VQDN * 4);
  unsigned short* vq_bf  = (unsigned short*)alloc((size_t)BBN * TTN * VQDN * 2);
  float* dots            = (float*)alloc((size_t)BBN * TTN * NEMBN * 4);
  unsigned short* h_bf   = (unsigned short*)alloc((size_t)BBN * HIDN * 2);
  unsigned* sync_cnt     = (unsigned*)alloc(256);
  float* diff_acc        = (float*)alloc(256);

  const int thr = 256;
  size_t nx = (size_t)BBN * TTN * EMBN;
  embed_kernel<<<(unsigned)((nx + thr - 1) / thr), thr, 0, stream>>>(
      inputs, embed_w, x_bf, nx);
  size_t n1 = (size_t)4 * HIDN * EMBN;
  f32_to_bf16_kernel<<<(unsigned)((n1 + thr - 1) / thr), thr, 0, stream>>>(
      w_ih, wih_bf, n1);
  size_t n2 = (size_t)4 * HIDN * HIDN;
  f32_to_bf16_kernel<<<(unsigned)((n2 + thr - 1) / thr), thr, 0, stream>>>(
      w_hh, whh_bf, n2);
  size_t n3 = (size_t)VQDN * HIDN;
  f32_to_bf16_kernel<<<(unsigned)((n3 + thr - 1) / thr), thr, 0, stream>>>(
      lin_w, lin_bf, n3);
  bias_kernel<<<(4 * HIDN + thr - 1) / thr, thr, 0, stream>>>(b_ih, b_hh, bias);
  size_t n4 = (size_t)NEMBN * VQDN;
  cbT_kernel<<<(unsigned)((n4 + thr - 1) / thr), thr, 0, stream>>>(codebook,
                                                                   cbT_bf);
  cbnorm_kernel<<<(NEMBN + thr - 1) / thr, thr, 0, stream>>>(codebook, cbnorm);
  init_kernel<<<(BBN * HIDN + thr - 1) / thr, thr, 0, stream>>>(h_bf, sync_cnt,
                                                                diff_acc);

  // xp = x @ w_ih^T + (b_ih + b_hh)   [16384 x 4096], K=512
  dim3 g1(4 * HIDN / 128, BBN * TTN / 64);
  wmma_gemm_tdm<<<g1, 256, 0, stream>>>(x_bf, wih_bf, xp, nullptr, bias,
                                        BBN * TTN, 4 * HIDN, EMBN, EMBN, EMBN,
                                        4 * HIDN);
  // persistent LSTM over 512 steps
  lstm_kernel<<<NWG, 256, 0, stream>>>(xp, whh_bf, h_bf, hs_bf, sync_cnt);
  // vq_states = hs @ lin_w^T   [16384 x 512], K=1024 (+ bf16 copy)
  dim3 g2(VQDN / 128, BBN * TTN / 64);
  wmma_gemm_tdm<<<g2, 256, 0, stream>>>(hs_bf, lin_bf, vqs, vq_bf, nullptr,
                                        BBN * TTN, VQDN, HIDN, HIDN, HIDN,
                                        VQDN);
  // dots = vq_states @ codebook  [16384 x 1024], K=512
  dim3 g3(NEMBN / 128, BBN * TTN / 64);
  wmma_gemm_tdm<<<g3, 256, 0, stream>>>(vq_bf, cbT_bf, dots, nullptr, nullptr,
                                        BBN * TTN, NEMBN, VQDN, VQDN, VQDN,
                                        NEMBN);
  // argmin + gather quantize into d_out + diff accumulation
  vq_argmin_kernel<<<BBN * TTN / 8, 256, 0, stream>>>(dots, cbnorm, vqs,
                                                      codebook, out, diff_acc);
  finalize_kernel<<<1, 32, 0, stream>>>(diff_acc, out);
}
